// CustomMultiheadAttention_27925877358901
// MI455X (gfx1250) — matrix operand order repaired and checked by reading, not compiled, not run on hardware
//
#include <hip/hip_runtime.h>

#define SEQ   2048
#define BATCH 2
#define EMB   1024
#define NH    16
#define DH    64
#define E3    3072
#define SCL   0.125f   // 1/sqrt(64)

typedef __attribute__((ext_vector_type(16))) __bf16 v16bf;
typedef __attribute__((ext_vector_type(8)))  __bf16 v8bf;
typedef __attribute__((ext_vector_type(8)))  float  v8f;
typedef __attribute__((ext_vector_type(4)))  int    v4i;

static __device__ __forceinline__ unsigned short f2bfbits(float f) {
  __bf16 h = (__bf16)f;
  return __builtin_bit_cast(unsigned short, h);
}
static __device__ __forceinline__ __bf16 bits2bf(unsigned short u) {
  return __builtin_bit_cast(__bf16, u);
}
static __device__ __forceinline__ v8f wmma_bf16(v16bf a, v16bf b, v8f c) {
  return __builtin_amdgcn_wmma_f32_16x16x32_bf16(false, a, false, b, (short)0, c,
                                                 false, false);
}

// ---- CDNA5 async global->LDS copy (ASYNCcnt) -------------------------------
// Probe round 2 revealed the builtin signature: (AS1 v4i*, AS3 v4i*, i32, i32)
#if __has_builtin(__builtin_amdgcn_global_load_async_to_lds_b128)
#define ASYNC_COPY_B128(gptr, lptr)                                            \
  __builtin_amdgcn_global_load_async_to_lds_b128(                              \
      (__attribute__((address_space(1))) v4i*)(gptr),                          \
      (__attribute__((address_space(3))) v4i*)(lptr), 0, 0)
#else
#define ASYNC_COPY_B128(gptr, lptr)                                            \
  do {                                                                         \
    unsigned _l = (unsigned)(uintptr_t)(lptr);                                 \
    unsigned long long _g = (unsigned long long)(uintptr_t)(gptr);             \
    asm volatile("global_load_async_to_lds_b128 %0, %1, off"                   \
                 :: "v"(_l), "v"(_g) : "memory");                              \
  } while (0)
#endif

#if __has_builtin(__builtin_amdgcn_s_wait_asynccnt)
#define WAIT_ASYNC(n) __builtin_amdgcn_s_wait_asynccnt(n)
#else
#define WAIT_ASYNC(n) asm volatile("s_wait_asynccnt %0" :: "i"(n) : "memory")
#endif

// ---------------------------------------------------------------------------
// Kernel 0: zero attn_mean accumulator (atomic-accumulated in kernel 2)
// ---------------------------------------------------------------------------
__global__ void zero_kernel(float* __restrict__ p, int n) {
  int i = blockIdx.x * blockDim.x + threadIdx.x;
  if (i < n) p[i] = 0.0f;
}

// ---------------------------------------------------------------------------
// Kernel 1: QKV projection GEMM  (M=B*S=4096, N=3E=3072, K=E=1024)
//   Register-blocked: one wave -> 16x64 output strip (A-frag reused 4x).
//   Q,K stored bf16 [B,H,S,d]; V stored transposed bf16 [B,H,d,S].
// ---------------------------------------------------------------------------
__global__ void qkv_kernel(const float* __restrict__ x,
                           const float* __restrict__ qkv_w,
                           const float* __restrict__ qkv_b,
                           unsigned short* __restrict__ q,
                           unsigned short* __restrict__ k,
                           unsigned short* __restrict__ vT) {
  const int lane = threadIdx.x & 31;
  const int wid  = threadIdx.x >> 5;
  const int tile = blockIdx.x * 4 + wid;         // 256 * 48 tiles
  const int tm   = tile / (E3 / 64);
  const int tng  = tile % (E3 / 64);             // 64-wide N group
  const int kg   = lane >> 4;
  const int col  = lane & 15;

  const int ar = tm * 16 + col;                  // GEMM row r = b*S + s
  const int ab = ar / SEQ;
  const int as = ar % SEQ;
  const float* xp = x + ((size_t)as * BATCH + ab) * EMB;

  v8f c[4] = {};
  for (int k0 = 0; k0 < EMB; k0 += 32) {
    v8f x0 = *(const v8f*)(xp + k0 + kg * 16);
    v8f x1 = *(const v8f*)(xp + k0 + kg * 16 + 8);
    v16bf a;
#pragma unroll
    for (int e = 0; e < 8; ++e) { a[e] = (__bf16)x0[e]; a[e + 8] = (__bf16)x1[e]; }
#pragma unroll
    for (int nt = 0; nt < 4; ++nt) {
      const float* wp = qkv_w + (size_t)(tng * 64 + nt * 16 + col) * EMB;
      v8f w0 = *(const v8f*)(wp + k0 + kg * 16);
      v8f w1 = *(const v8f*)(wp + k0 + kg * 16 + 8);
      v16bf bm;
#pragma unroll
      for (int e = 0; e < 8; ++e) { bm[e] = (__bf16)w0[e]; bm[e + 8] = (__bf16)w1[e]; }
      c[nt] = wmma_bf16(a, bm, c[nt]);
    }
  }

#pragma unroll
  for (int nt = 0; nt < 4; ++nt) {
    const int n    = tng * 64 + nt * 16 + col;
    const float bias = qkv_b[n];
    const int t   = n / EMB;                     // 0=q 1=k 2=v (const per tile)
    const int rem = n % EMB;
    const int h   = rem / DH;
    const int dd  = rem % DH;
#pragma unroll
    for (int i = 0; i < 8; ++i) {
      const int rr = tm * 16 + i + 8 * kg;       // C layout: M = i + 8*(lane>=16)
      const int bb = rr / SEQ;
      const int ss = rr % SEQ;
      const unsigned short val = f2bfbits(c[nt][i] + bias);
      if (t == 0)      q[(((size_t)bb * NH + h) * SEQ + ss) * DH + dd] = val;
      else if (t == 1) k[(((size_t)bb * NH + h) * SEQ + ss) * DH + dd] = val;
      else             vT[(((size_t)bb * NH + h) * DH + dd) * SEQ + ss] = val;
    }
  }
}

// ---------------------------------------------------------------------------
// Kernel 2: attention. 4 waves/block share (b,h); K and V 32-key blocks are
// staged cooperatively into LDS with GLOBAL_LOAD_ASYNC_TO_LDS_B128, double
// buffered and drained with s_wait_asynccnt. Two-pass flash softmax; P is
// re-laid out C->A fragment through LDS; attn_mean via f32 global atomics.
// ---------------------------------------------------------------------------
static __device__ __forceinline__ void stage_k(const unsigned short* kh,
                                               unsigned short* kl,
                                               int kb, int tid) {
  // 32 keys x 64 depth bf16 = 4KB contiguous; 128 thr x 2 x 16B
  const unsigned short* g = kh + (size_t)kb * 32 * DH;
#pragma unroll
  for (int t = 0; t < 2; ++t)
    ASYNC_COPY_B128(g + t * 1024 + tid * 8, kl + t * 1024 + tid * 8);
}
static __device__ __forceinline__ void stage_v(const unsigned short* vh,
                                               unsigned short* vl,
                                               int kb, int tid) {
  // 64 depth-rows x 32 keys bf16 (row stride S in global) = 4KB
#pragma unroll
  for (int t = 0; t < 2; ++t) {
    const int chunk = tid * 2 + t;               // 0..255 16B chunks
    const int row  = chunk >> 2;                 // depth row 0..63
    const int part = chunk & 3;                  // 4 x 16B per row
    ASYNC_COPY_B128(vh + (size_t)row * SEQ + kb * 32 + part * 8,
                    vl + row * 32 + part * 8);
  }
}

__global__ void attn_kernel(const unsigned short* __restrict__ q,
                            const unsigned short* __restrict__ k,
                            const unsigned short* __restrict__ vT,
                            const float* __restrict__ attn_mask,
                            const unsigned char* __restrict__ kpm,
                            unsigned short* __restrict__ wa,
                            float* __restrict__ amean) {
  __shared__ __align__(64) unsigned short kbuf[2][32 * DH];   // 2 x 4KB
  __shared__ __align__(64) unsigned short vbuf[2][DH * 32];   // 2 x 4KB
  __shared__ __align__(64) unsigned short pshare[4][16 * 32]; // 4 x 1KB

  const int tid  = threadIdx.x;
  const int lane = tid & 31;
  const int wid  = tid >> 5;
  const int unit = blockIdx.x * 4 + wid;
  const int b    = unit / (NH * (SEQ / 16));
  const int rest = unit % (NH * (SEQ / 16));
  const int h    = rest / (SEQ / 16);            // same for all 4 waves: 4|128
  const int qt   = rest % (SEQ / 16);
  const int kg   = lane >> 4;
  const int col  = lane & 15;

  const unsigned short* qh = q  + ((size_t)(b * NH + h) * SEQ) * DH;
  const unsigned short* kh = k  + ((size_t)(b * NH + h) * SEQ) * DH;
  const unsigned short* vh = vT + ((size_t)(b * NH + h) * DH) * SEQ;

  // Q A-fragments (K = 0..31 and 32..63), resident for the whole kernel
  v16bf aq0, aq1;
  {
    const unsigned short* qr = qh + (size_t)(qt * 16 + col) * DH;
    v8bf a00 = *(const v8bf*)(qr + kg * 16);
    v8bf a01 = *(const v8bf*)(qr + kg * 16 + 8);
    v8bf a10 = *(const v8bf*)(qr + 32 + kg * 16);
    v8bf a11 = *(const v8bf*)(qr + 32 + kg * 16 + 8);
#pragma unroll
    for (int e = 0; e < 8; ++e) {
      aq0[e] = a00[e]; aq0[e + 8] = a01[e];
      aq1[e] = a10[e]; aq1[e + 8] = a11[e];
    }
  }

  float m[8], l[8];
#pragma unroll
  for (int i = 0; i < 8; ++i) { m[i] = -INFINITY; l[i] = 0.0f; }

  // ---- Pass 1: online row max/sum, K double-buffered through LDS ----
  stage_k(kh, kbuf[0], 0, tid);
  for (int kb = 0; kb < SEQ / 32; ++kb) {
    if (kb + 1 < SEQ / 32) { stage_k(kh, kbuf[(kb + 1) & 1], kb + 1, tid); WAIT_ASYNC(2); }
    else                   { WAIT_ASYNC(0); }
    __syncthreads();                              // block kb resident in LDS
    const unsigned short* kcur = kbuf[kb & 1];
#pragma unroll
    for (int half = 0; half < 2; ++half) {
      const int keybase = kb * 32 + half * 16;
      const unsigned short* kr = kcur + (half * 16 + col) * DH + kg * 16;
      v16bf b0 = *(const v16bf*)(kr);
      v16bf b1 = *(const v16bf*)(kr + 32);
      v8f c = {};
      c = wmma_bf16(aq0, b0, c);
      c = wmma_bf16(aq1, b1, c);
      const bool pad = kpm[b * SEQ + keybase + col] != 0;
#pragma unroll
      for (int i = 0; i < 8; ++i) {
        const int qr = qt * 16 + i + 8 * kg;
        float s = c[i] * SCL + attn_mask[(size_t)qr * SEQ + keybase + col];
        if (pad) s = -INFINITY;
        float mn = fmaxf(m[i], s);
        l[i] = l[i] * __expf(m[i] - mn) + __expf(s - mn);
        m[i] = mn;
      }
    }
    __syncthreads();                              // all reads done before overwrite
  }

  // Merge (m,l) across the 16 lanes sharing each row (xor<16 stays in half)
#pragma unroll
  for (int i = 0; i < 8; ++i) {
#pragma unroll
    for (int off = 1; off < 16; off <<= 1) {
      float mo = __shfl_xor(m[i], off, 32);
      float lo = __shfl_xor(l[i], off, 32);
      float mn = fmaxf(m[i], mo);
      l[i] = l[i] * __expf(m[i] - mn) + lo * __expf(mo - mn);
      m[i] = mn;
    }
    l[i] = 1.0f / l[i];                           // reciprocal for pass 2
  }

  // ---- Pass 2: P -> attn_mean atomics + P@V, K and V async double-buffered
  v8f o0 = {}, o1 = {}, o2 = {}, o3 = {};
  const float invH = 1.0f / NH;
  stage_k(kh, kbuf[0], 0, tid);
  stage_v(vh, vbuf[0], 0, tid);
  for (int kb = 0; kb < SEQ / 32; ++kb) {
    if (kb + 1 < SEQ / 32) {
      stage_k(kh, kbuf[(kb + 1) & 1], kb + 1, tid);
      stage_v(vh, vbuf[(kb + 1) & 1], kb + 1, tid);
      WAIT_ASYNC(4);
    } else {
      WAIT_ASYNC(0);
    }
    __syncthreads();
    const unsigned short* kcur = kbuf[kb & 1];
    const unsigned short* vcur = vbuf[kb & 1];
#pragma unroll
    for (int half = 0; half < 2; ++half) {
      const int keybase = kb * 32 + half * 16;
      const unsigned short* kr = kcur + (half * 16 + col) * DH + kg * 16;
      v16bf b0 = *(const v16bf*)(kr);
      v16bf b1 = *(const v16bf*)(kr + 32);
      v8f c = {};
      c = wmma_bf16(aq0, b0, c);
      c = wmma_bf16(aq1, b1, c);
      const bool pad = kpm[b * SEQ + keybase + col] != 0;
#pragma unroll
      for (int i = 0; i < 8; ++i) {
        const int qr = qt * 16 + i + 8 * kg;
        float s = c[i] * SCL + attn_mask[(size_t)qr * SEQ + keybase + col];
        if (pad) s = -INFINITY;
        float p = __expf(s - m[i]) * l[i];
        atomicAdd(&amean[((size_t)b * SEQ + qr) * SEQ + keybase + col], p * invH);
        pshare[wid][(i + 8 * kg) * 32 + half * 16 + col] = f2bfbits(p);
      }
    }
    __syncthreads();                              // P tile visible wave-wide
    v16bf pa;                                     // C layout -> A fragment
#pragma unroll
    for (int e = 0; e < 16; ++e) {
      const int kk = kg * 16 + e;
      pa[e] = bits2bf(pshare[wid][col * 32 + kk]);
    }
    v16bf bv0 = *(const v16bf*)(vcur + (0 * 16 + col) * 32 + kg * 16);
    v16bf bv1 = *(const v16bf*)(vcur + (1 * 16 + col) * 32 + kg * 16);
    v16bf bv2 = *(const v16bf*)(vcur + (2 * 16 + col) * 32 + kg * 16);
    v16bf bv3 = *(const v16bf*)(vcur + (3 * 16 + col) * 32 + kg * 16);
    o0 = wmma_bf16(pa, bv0, o0);
    o1 = wmma_bf16(pa, bv1, o1);
    o2 = wmma_bf16(pa, bv2, o2);
    o3 = wmma_bf16(pa, bv3, o3);
    __syncthreads();                              // done reading before overwrite
  }

  // Store attention output strip into wa [B*S, E] bf16 (row r = b*S+s)
#pragma unroll
  for (int i = 0; i < 8; ++i) {
    const int ss = qt * 16 + i + 8 * kg;
    unsigned short* wrow = wa + ((size_t)b * SEQ + ss) * EMB + h * DH;
    wrow[0 * 16 + col] = f2bfbits(o0[i]);
    wrow[1 * 16 + col] = f2bfbits(o1[i]);
    wrow[2 * 16 + col] = f2bfbits(o2[i]);
    wrow[3 * 16 + col] = f2bfbits(o3[i]);
  }
}

// ---------------------------------------------------------------------------
// Kernel 3: output projection (M=4096, N=1024, K=1024), 16x64 per wave,
// scatter to [S,B,E]
// ---------------------------------------------------------------------------
__global__ void proj_kernel(const unsigned short* __restrict__ wa,
                            const float* __restrict__ out_w,
                            const float* __restrict__ out_b,
                            float* __restrict__ out) {
  const int lane = threadIdx.x & 31;
  const int wid  = threadIdx.x >> 5;
  const int tile = blockIdx.x * 4 + wid;          // 256 * 16 tiles
  const int tm   = tile / (EMB / 64);
  const int tng  = tile % (EMB / 64);
  const int kg   = lane >> 4;
  const int col  = lane & 15;

  const unsigned short* ap = wa + (size_t)(tm * 16 + col) * EMB;

  v8f c[4] = {};
  for (int k0 = 0; k0 < EMB; k0 += 32) {
    v8bf a0 = *(const v8bf*)(ap + k0 + kg * 16);
    v8bf a1 = *(const v8bf*)(ap + k0 + kg * 16 + 8);
    v16bf a;
#pragma unroll
    for (int e = 0; e < 8; ++e) { a[e] = a0[e]; a[e + 8] = a1[e]; }
#pragma unroll
    for (int nt = 0; nt < 4; ++nt) {
      const float* wp = out_w + (size_t)(tng * 64 + nt * 16 + col) * EMB;
      v8f w0 = *(const v8f*)(wp + k0 + kg * 16);
      v8f w1 = *(const v8f*)(wp + k0 + kg * 16 + 8);
      v16bf bm;
#pragma unroll
      for (int e = 0; e < 8; ++e) { bm[e] = (__bf16)w0[e]; bm[e + 8] = (__bf16)w1[e]; }
      c[nt] = wmma_bf16(a, bm, c[nt]);
    }
  }

#pragma unroll
  for (int nt = 0; nt < 4; ++nt) {
    const int n = tng * 64 + nt * 16 + col;
    const float bias = out_b[n];
#pragma unroll
    for (int i = 0; i < 8; ++i) {
      const int rr = tm * 16 + i + 8 * kg;        // r = b*S + s
      const int bb = rr / SEQ;
      const int ss = rr % SEQ;
      out[((size_t)ss * BATCH + bb) * EMB + n] = c[nt][i] + bias;
    }
  }
}

// ---------------------------------------------------------------------------
extern "C" void kernel_launch(void* const* d_in, const int* in_sizes, int n_in,
                              void* d_out, int out_size, void* d_ws, size_t ws_size,
                              hipStream_t stream) {
  const float*         x         = (const float*)d_in[0];
  const float*         attn_mask = (const float*)d_in[1];
  const unsigned char* kpm       = (const unsigned char*)d_in[2];
  const float*         qkv_w     = (const float*)d_in[3];
  const float*         qkv_b     = (const float*)d_in[4];
  const float*         out_w     = (const float*)d_in[5];
  const float*         out_b     = (const float*)d_in[6];

  float* out   = (float*)d_out;
  float* amean = out + (size_t)SEQ * BATCH * EMB;   // [B,S,S] section of d_out

  const size_t QKV_ELEMS = (size_t)BATCH * NH * SEQ * DH;   // 4M bf16 each
  unsigned short* q  = (unsigned short*)d_ws;
  unsigned short* k  = q  + QKV_ELEMS;
  unsigned short* vT = k  + QKV_ELEMS;
  unsigned short* wa = vT + QKV_ELEMS;              // [B*S, E] bf16

  const int amean_n = BATCH * SEQ * SEQ;
  zero_kernel<<<(amean_n + 255) / 256, 256, 0, stream>>>(amean, amean_n);

  // 256 M-tiles x 48 N-groups, 4 waves per 128-thread block
  qkv_kernel<<<(256 * 48) / 4, 128, 0, stream>>>(x, qkv_w, qkv_b, q, k, vT);

  // B*H*(S/16) = 4096 wave-units, 4 per block (block shares b,h)
  attn_kernel<<<(BATCH * NH * (SEQ / 16)) / 4, 128, 0, stream>>>(
      q, k, vT, attn_mask, kpm, wa, amean);

  // 256 M-tiles x 16 N-groups, 4 per block
  proj_kernel<<<(256 * 16) / 4, 128, 0, stream>>>(wa, out_w, out_b, out);
}
